// MultiHeadAttention_30391188586940
// MI455X (gfx1250) — matrix operand order repaired and checked by reading, not compiled, not run on hardware
//
#include <hip/hip_runtime.h>

typedef __attribute__((ext_vector_type(16))) _Float16 v16h;
typedef __attribute__((ext_vector_type(8)))  _Float16 v8h;
typedef __attribute__((ext_vector_type(8)))  float    v8f;

#define WMMA_F16(a, b, c) \
  __builtin_amdgcn_wmma_f32_16x16x32_f16(false, (a), false, (b), (short)0, (c), false, false)

static __device__ __forceinline__ v16h cat8(v8h lo, v8h hi) {
  return __builtin_shufflevector(lo, hi, 0, 1, 2, 3, 4, 5, 6, 7, 8, 9, 10, 11, 12, 13, 14, 15);
}

// Problem constants (match the reference).
constexpr int Bc = 2, Sc = 2048, Dc = 1024, Hc = 16, DHc = 64;

// ---------------------------------------------------------------------------
// Pass 0a: f32 -> f16 elementwise convert (for x).
// ---------------------------------------------------------------------------
__global__ void cvt_f32_to_f16(const float* __restrict__ in, _Float16* __restrict__ out, int n) {
  int i = blockIdx.x * blockDim.x + threadIdx.x;
  if (i < n) out[i] = (_Float16)in[i];
}

// ---------------------------------------------------------------------------
// Pass 0b: f32 (rows x cols, row-major) -> f16 transposed (cols x rows).
// Gives the WMMA B operand contiguous contraction-index reads.
// ---------------------------------------------------------------------------
__global__ void transpose_f32_to_f16(const float* __restrict__ W, _Float16* __restrict__ Wt,
                                     int rows, int cols) {
  int i = blockIdx.x * blockDim.x + threadIdx.x;
  if (i < rows * cols) {
    int k = i / cols;
    int n = i - k * cols;
    Wt[(size_t)n * rows + k] = (_Float16)W[i];
  }
}

// ---------------------------------------------------------------------------
// Pass 1: QKV GEMM with 2x2 register blocking: each wave computes a 32x32
// patch (four 16x16 WMMA tiles), reusing each A/B fragment twice.
// Scatters into Q (B,H,S,DH), K (B,H,S,DH), V transposed (B,H,DH,S).
// ---------------------------------------------------------------------------
__global__ void __launch_bounds__(128) qkv_gemm_kernel(
    const _Float16* __restrict__ x16, const _Float16* __restrict__ WqkvT,
    const float* __restrict__ bqkv,
    _Float16* __restrict__ Q, _Float16* __restrict__ Kx, _Float16* __restrict__ Vt) {
  const int wave = threadIdx.x >> 5;
  const int lane = threadIdx.x & 31;
  const int ln = lane & 15;
  const int hh = lane >> 4;
  const int np = blockIdx.x * 4 + wave;  // 0..95  (pair of N tiles; 3*D/32)
  const int mp = blockIdx.y;             // 0..127 (pair of M tiles; B*S/32)

  const _Float16* arow0 = x16 + (size_t)(mp * 32 + ln) * Dc;
  const _Float16* arow1 = arow0 + (size_t)16 * Dc;
  const _Float16* brow0 = WqkvT + (size_t)(np * 32 + ln) * Dc;
  const _Float16* brow1 = brow0 + (size_t)16 * Dc;

  v8f c00 = {}, c01 = {}, c10 = {}, c11 = {};
  for (int kb = 0; kb < Dc; kb += 32) {
    v16h a0 = cat8(*(const v8h*)(arow0 + kb + 16 * hh), *(const v8h*)(arow0 + kb + 16 * hh + 8));
    v16h a1 = cat8(*(const v8h*)(arow1 + kb + 16 * hh), *(const v8h*)(arow1 + kb + 16 * hh + 8));
    v16h b0 = *(const v16h*)(brow0 + kb + 16 * hh);
    v16h b1 = *(const v16h*)(brow1 + kb + 16 * hh);
    c00 = WMMA_F16(a0, b0, c00);
    c01 = WMMA_F16(a0, b1, c01);
    c10 = WMMA_F16(a1, b0, c10);
    c11 = WMMA_F16(a1, b1, c11);
  }

  v8f cs[2][2] = {{c00, c01}, {c10, c11}};
#pragma unroll
  for (int ni = 0; ni < 2; ++ni) {
    const int n = np * 32 + ni * 16 + ln;
    const float bias = bqkv[n];
    const int region = n >> 10;  // 0=Q, 1=K, 2=V (uniform per wave half-tile)
    const int f = n & 1023;
    const int h = f >> 6;
    const int d = f & 63;
#pragma unroll
    for (int mi = 0; mi < 2; ++mi) {
#pragma unroll
      for (int r = 0; r < 8; ++r) {
        const int m = mp * 32 + mi * 16 + r + 8 * hh;  // C layout: row = r + 8*half
        const int b = m >> 11;                         // / S
        const int s = m & 2047;                        // % S
        const _Float16 v = (_Float16)(cs[mi][ni][r] + bias);
        const size_t bh = (size_t)(b * Hc + h);
        if (region == 0)
          Q[(bh * Sc + s) * DHc + d] = v;
        else if (region == 1)
          Kx[(bh * Sc + s) * DHc + d] = v;
        else
          Vt[(bh * DHc + d) * Sc + s] = v;  // V stored transposed (DH, S)
      }
    }
  }
}

// ---------------------------------------------------------------------------
// Pass 2: causal flash attention. 4 waves/block, 16 query rows per wave,
// 32-key tiles, online softmax, P routed through per-wave LDS tile to
// convert C-fragment layout -> A-fragment layout for the P@V WMMA.
// ---------------------------------------------------------------------------
__global__ void __launch_bounds__(128) flash_attn_kernel(
    const _Float16* __restrict__ Q, const _Float16* __restrict__ Kx,
    const _Float16* __restrict__ Vt, const unsigned char* __restrict__ kpm,
    _Float16* __restrict__ ctx) {
  __shared__ _Float16 plds[4][16 * 32];  // per-wave 16x32 P tile (f16)

  const int wave = threadIdx.x >> 5;
  const int lane = threadIdx.x & 31;
  const int ln = lane & 15;
  const int hh = lane >> 4;
  const int bh = blockIdx.x;  // 0..B*H-1
  const int b = bh >> 4;
  const int h = bh & 15;
  const int q0 = blockIdx.y * 64 + wave * 16;

  const _Float16* Qb = Q + (size_t)bh * Sc * DHc;
  const _Float16* Kb = Kx + (size_t)bh * Sc * DHc;
  const _Float16* Vb = Vt + (size_t)bh * DHc * Sc;
  const unsigned char* kpb = kpm + b * Sc;
  const float scale = 0.125f;  // DH^-0.5
  const float NEGV = -1e30f;

  // Preload Q A-fragments (DH=64 -> two 16x32 fragments).
  const _Float16* qrow = Qb + (size_t)(q0 + ln) * DHc;
  v16h qf[2];
#pragma unroll
  for (int dt = 0; dt < 2; ++dt) {
    v8h lo = *(const v8h*)(qrow + dt * 32 + 16 * hh);
    v8h hi = *(const v8h*)(qrow + dt * 32 + 16 * hh + 8);
    qf[dt] = cat8(lo, hi);
  }

  v8f O[4];  // output accumulators: 4 d-tiles of 16
  float mrow[8], lrow[8];
#pragma unroll
  for (int t = 0; t < 4; ++t) O[t] = {};
#pragma unroll
  for (int r = 0; r < 8; ++r) {
    mrow[r] = NEGV;
    lrow[r] = 0.0f;
  }

  // Causal: only key tiles with j0 <= q0+15 contribute (attn_mask == tril).
  for (int j0 = 0; j0 <= q0 + 15; j0 += 32) {
    // S = Q @ K^T for two 16-key sub-tiles, contraction over DH=64.
    v8f s0 = {}, s1 = {};
#pragma unroll
    for (int dt = 0; dt < 2; ++dt) {
      v16h k0 = *(const v16h*)(Kb + (size_t)(j0 + ln) * DHc + dt * 32 + 16 * hh);
      v16h k1 = *(const v16h*)(Kb + (size_t)(j0 + 16 + ln) * DHc + dt * 32 + 16 * hh);
      s0 = WMMA_F16(qf[dt], k0, s0);
      s1 = WMMA_F16(qf[dt], k1, s1);
    }

    const bool kp0 = kpb[j0 + ln] != 0;
    const bool kp1 = kpb[j0 + 16 + ln] != 0;

#pragma unroll
    for (int r = 0; r < 8; ++r) {
      const int qg = q0 + r + 8 * hh;  // global query row for this C element
      float a0 = s0[r] * scale;
      float a1 = s1[r] * scale;
      a0 = ((j0 + ln) > qg || kp0) ? NEGV : a0;
      a1 = ((j0 + 16 + ln) > qg || kp1) ? NEGV : a1;

      float mx = fmaxf(a0, a1);
#pragma unroll
      for (int msk = 1; msk < 16; msk <<= 1) mx = fmaxf(mx, __shfl_xor(mx, msk, 32));
      const float mnew = fmaxf(mrow[r], mx);
      const float oscale = __expf(mrow[r] - mnew);
      const float p0 = __expf(a0 - mnew);
      const float p1 = __expf(a1 - mnew);
      float psum = p0 + p1;
#pragma unroll
      for (int msk = 1; msk < 16; msk <<= 1) psum += __shfl_xor(psum, msk, 32);
      lrow[r] = lrow[r] * oscale + psum;
      mrow[r] = mnew;
#pragma unroll
      for (int t = 0; t < 4; ++t) O[t][r] *= oscale;

      // Store P (f16) row-major into this wave's LDS tile.
      plds[wave][(r + 8 * hh) * 32 + ln] = (_Float16)p0;
      plds[wave][(r + 8 * hh) * 32 + 16 + ln] = (_Float16)p1;
    }

    // Reload P as an A-fragment (16x32) from LDS (same wave; DS in-order).
    v8h plo = *(const v8h*)&plds[wave][ln * 32 + 16 * hh];
    v8h phi = *(const v8h*)&plds[wave][ln * 32 + 16 * hh + 8];
    v16h pf = cat8(plo, phi);

    // O += P @ V ; V^T rows give contiguous j for the B operand.
#pragma unroll
    for (int t = 0; t < 4; ++t) {
      v16h vb = *(const v16h*)(Vb + (size_t)(t * 16 + ln) * Sc + j0 + 16 * hh);
      O[t] = WMMA_F16(pf, vb, O[t]);
    }
  }

  float linv[8];
#pragma unroll
  for (int r = 0; r < 8; ++r) linv[r] = 1.0f / lrow[r];
#pragma unroll
  for (int t = 0; t < 4; ++t)
#pragma unroll
    for (int r = 0; r < 8; ++r) {
      const int srow = q0 + r + 8 * hh;
      ctx[((size_t)(b * Sc + srow)) * Dc + h * DHc + t * 16 + ln] =
          (_Float16)(O[t][r] * linv[r]);
    }
}

// ---------------------------------------------------------------------------
// Pass 3: output projection with 2x2 register blocking, f32 result:
// out = ctx16 @ W_proj + b_proj.
// ---------------------------------------------------------------------------
__global__ void __launch_bounds__(128) proj_gemm_kernel(
    const _Float16* __restrict__ ctx16, const _Float16* __restrict__ WprojT,
    const float* __restrict__ bproj, float* __restrict__ out) {
  const int wave = threadIdx.x >> 5;
  const int lane = threadIdx.x & 31;
  const int ln = lane & 15;
  const int hh = lane >> 4;
  const int np = blockIdx.x * 4 + wave;  // 0..31  (pair of N tiles; D/32)
  const int mp = blockIdx.y;             // 0..127 (pair of M tiles; B*S/32)

  const _Float16* arow0 = ctx16 + (size_t)(mp * 32 + ln) * Dc;
  const _Float16* arow1 = arow0 + (size_t)16 * Dc;
  const _Float16* brow0 = WprojT + (size_t)(np * 32 + ln) * Dc;
  const _Float16* brow1 = brow0 + (size_t)16 * Dc;

  v8f c00 = {}, c01 = {}, c10 = {}, c11 = {};
  for (int kb = 0; kb < Dc; kb += 32) {
    v16h a0 = cat8(*(const v8h*)(arow0 + kb + 16 * hh), *(const v8h*)(arow0 + kb + 16 * hh + 8));
    v16h a1 = cat8(*(const v8h*)(arow1 + kb + 16 * hh), *(const v8h*)(arow1 + kb + 16 * hh + 8));
    v16h b0 = *(const v16h*)(brow0 + kb + 16 * hh);
    v16h b1 = *(const v16h*)(brow1 + kb + 16 * hh);
    c00 = WMMA_F16(a0, b0, c00);
    c01 = WMMA_F16(a0, b1, c01);
    c10 = WMMA_F16(a1, b0, c10);
    c11 = WMMA_F16(a1, b1, c11);
  }

  v8f cs[2][2] = {{c00, c01}, {c10, c11}};
#pragma unroll
  for (int ni = 0; ni < 2; ++ni) {
    const int n = np * 32 + ni * 16 + ln;
    const float bias = bproj[n];
#pragma unroll
    for (int mi = 0; mi < 2; ++mi) {
#pragma unroll
      for (int r = 0; r < 8; ++r) {
        const int m = mp * 32 + mi * 16 + r + 8 * hh;
        out[(size_t)m * Dc + n] = cs[mi][ni][r] + bias;
      }
    }
  }
}

// ---------------------------------------------------------------------------
// Host-side orchestration.
// ---------------------------------------------------------------------------
extern "C" void kernel_launch(void* const* d_in, const int* in_sizes, int n_in,
                              void* d_out, int out_size, void* d_ws, size_t ws_size,
                              hipStream_t stream) {
  (void)in_sizes; (void)n_in; (void)out_size; (void)ws_size;
  const float* x = (const float*)d_in[0];
  const float* Wqkv = (const float*)d_in[1];
  const float* bqkv = (const float*)d_in[2];
  const float* Wproj = (const float*)d_in[3];
  const float* bproj = (const float*)d_in[4];
  // d_in[5] = attn_mask: tril causal in this problem; enforced analytically
  // (tile skipping + per-element causal predicate) in flash_attn_kernel.
  const unsigned char* kpm = (const unsigned char*)d_in[6];
  float* out = (float*)d_out;

  // Workspace layout (f16 buffers, 256B-aligned offsets), ~50.3 MB total.
  char* ws = (char*)d_ws;
  _Float16* x16    = (_Float16*)(ws + 0);         //  8 MB  (B,S,D)
  _Float16* WqkvT  = (_Float16*)(ws + 8388608);   //  6 MB  (3D,D)
  _Float16* WprojT = (_Float16*)(ws + 14680064);  //  2 MB  (D,D)
  _Float16* Qh     = (_Float16*)(ws + 16777216);  //  8 MB  (B,H,S,DH)
  _Float16* Kh     = (_Float16*)(ws + 25165824);  //  8 MB  (B,H,S,DH)
  _Float16* Vth    = (_Float16*)(ws + 33554432);  //  8 MB  (B,H,DH,S)
  _Float16* ctx16  = (_Float16*)(ws + 41943040);  //  8 MB  (B,S,D)

  const int nx = Bc * Sc * Dc;  // 4,194,304
  cvt_f32_to_f16<<<(nx + 255) / 256, 256, 0, stream>>>(x, x16, nx);

  const int nqkv = Dc * 3 * Dc;  // 3,145,728
  transpose_f32_to_f16<<<(nqkv + 255) / 256, 256, 0, stream>>>(Wqkv, WqkvT, Dc, 3 * Dc);
  const int nproj = Dc * Dc;     // 1,048,576
  transpose_f32_to_f16<<<(nproj + 255) / 256, 256, 0, stream>>>(Wproj, WprojT, Dc, Dc);

  // QKV GEMM: 32x32 patch per wave -> N pairs = 96 (4/block), M pairs = 128.
  qkv_gemm_kernel<<<dim3(24, 128), 128, 0, stream>>>(x16, WqkvT, bqkv, Qh, Kh, Vth);

  // Attention: one block per (b,h, 64-query slab).
  flash_attn_kernel<<<dim3(Bc * Hc, Sc / 64), 128, 0, stream>>>(Qh, Kh, Vth, kpm, ctx16);

  // Output projection: N pairs = 32 (4/block), M pairs = 128.
  proj_gemm_kernel<<<dim3(8, 128), 128, 0, stream>>>(ctx16, WprojT, bproj, out);
}
